// GnnEmbedder_54116587929912
// MI455X (gfx1250) — matrix operand order repaired and checked by reading, not compiled, not run on hardware
//
#include <hip/hip_runtime.h>

// ---------------------------------------------------------------------------
// GNN embedder (TwoPartEmbedding + 2x SAGEConv mean) for MI455X / gfx1250.
// GEMMs use v_wmma_f32_16x16x32_f16 (f16 A/B, f32 accum). Edge aggregation
// uses f16 features (L2-resident, 25.6MB) + f32 global atomics.
// ---------------------------------------------------------------------------

#define N_CLIENTS 1000000
#define N_ITEMS   100000
#define D_FEAT    128
#define N_NODES   100000
#define N_EDGES   1600000
#define M_TILES   (N_NODES / 16)   // 6250, exact

typedef __attribute__((ext_vector_type(16))) _Float16 v16h;
typedef __attribute__((ext_vector_type(8)))  _Float16 v8h;
typedef __attribute__((ext_vector_type(4)))  _Float16 v4h;
typedef __attribute__((ext_vector_type(8)))  float    v8f;
typedef __attribute__((ext_vector_type(4)))  float    v4f;

// ---------------------------------------------------------------------------
// Repack four 128x128 f32 weights into WMMA B-fragment order (f16).
// Layout: wblob[w][kt(4)][nt(8)][lane(32)][i(16)]  (halfs)
//   lane 0-15 : K = kt*32 + i       , N = nt*16 + lane
//   lane16-31 : K = kt*32 + 16 + i  , N = nt*16 + (lane-16)
// ---------------------------------------------------------------------------
__global__ __launch_bounds__(256) void prep_w_kernel(
    const float* __restrict__ Ws1, const float* __restrict__ Wn1,
    const float* __restrict__ Ws2, const float* __restrict__ Wn2,
    _Float16* __restrict__ wblob)
{
    int tid = blockIdx.x * 256 + threadIdx.x;        // 4 * 16384 threads
    const float* W = (tid < 16384) ? Ws1 : (tid < 32768) ? Wn1
                   : (tid < 49152) ? Ws2 : Wn2;
    int r    = tid & 16383;
    int frag = r >> 9;            // 0..31  (kt*8 + nt)
    int lane = (r >> 4) & 31;
    int i    = r & 15;
    int kt   = frag >> 3;
    int nt   = frag & 7;
    int k    = kt * 32 + ((lane >> 4) << 4) + i;
    int n    = nt * 16 + (lane & 15);
    wblob[tid] = (_Float16)W[k * 128 + n];
}

// ---------------------------------------------------------------------------
// Two-part embedding gather -> f16 features. Also zero agg + deg.
// tid = node*32 + t ; t covers dims [4t, 4t+3].
// ---------------------------------------------------------------------------
__global__ __launch_bounds__(256) void embed_kernel(
    const float* __restrict__ client_table, const float* __restrict__ item_table,
    const int* __restrict__ node_ids,
    _Float16* __restrict__ xh, float* __restrict__ agg, float* __restrict__ deg)
{
    int tid = blockIdx.x * 256 + threadIdx.x;
    int n = tid >> 5, t = tid & 31;
    if (n >= N_NODES) return;
    int id = node_ids[n];
    const float* srcp = (id < N_CLIENTS)
        ? client_table + (size_t)id * D_FEAT
        : item_table + (size_t)(id - N_CLIENTS) * D_FEAT;
    v4f v = *(const v4f*)(srcp + t * 4);
    v4h h;
#pragma unroll
    for (int i = 0; i < 4; ++i) h[i] = (_Float16)v[i];
    *(v4h*)(xh + n * D_FEAT + t * 4) = h;
    v4f z = {};
    *(v4f*)(agg + n * D_FEAT + t * 4) = z;
    if (t == 0) deg[n] = 0.0f;
}

// ---------------------------------------------------------------------------
// Degree: one thread per edge.
// ---------------------------------------------------------------------------
__global__ __launch_bounds__(256) void degree_kernel(
    const int* __restrict__ dst, float* __restrict__ deg)
{
    int e = blockIdx.x * 256 + threadIdx.x;
    if (e >= N_EDGES) return;
    atomicAdd(&deg[dst[e]], 1.0f);
}

// ---------------------------------------------------------------------------
// Segment-sum scatter: tid = edge*32 + t, 4 dims per thread.
// ---------------------------------------------------------------------------
__global__ __launch_bounds__(256) void scatter_kernel(
    const int* __restrict__ src, const int* __restrict__ dst,
    const _Float16* __restrict__ feat, float* __restrict__ agg)
{
    int tid = blockIdx.x * 256 + threadIdx.x;
    int e = tid >> 5, t = tid & 31;
    if (e >= N_EDGES) return;
    int s = src[e], d = dst[e];
    v4h v = *(const v4h*)(feat + s * D_FEAT + t * 4);
    float* ap = agg + d * D_FEAT + t * 4;
#pragma unroll
    for (int i = 0; i < 4; ++i) atomicAdd(ap + i, (float)v[i]);
}

// ---------------------------------------------------------------------------
// h_neigh = agg / max(deg,1) -> f16 ; re-zero agg for next layer.
// ---------------------------------------------------------------------------
__global__ __launch_bounds__(256) void finalize_kernel(
    float* __restrict__ agg, const float* __restrict__ deg,
    _Float16* __restrict__ hn)
{
    int tid = blockIdx.x * 256 + threadIdx.x;
    int n = tid >> 5, t = tid & 31;
    if (n >= N_NODES) return;
    float inv = 1.0f / fmaxf(deg[n], 1.0f);
    v4f a = *(v4f*)(agg + n * D_FEAT + t * 4);
    v4h h;
#pragma unroll
    for (int i = 0; i < 4; ++i) h[i] = (_Float16)(a[i] * inv);
    *(v4h*)(hn + n * D_FEAT + t * 4) = h;
    v4f z = {};
    *(v4f*)(agg + n * D_FEAT + t * 4) = z;
}

// ---------------------------------------------------------------------------
// Fused SAGE GEMM:  out = act( Aself @ Wself + Aneigh @ Wneigh + b )
// 8 waves / block, one 16x128 output tile per wave, K=128.
// Both weight blobs (2 x 32KB f16, fragment-packed) staged in LDS.
// 2 matrices * 4 ktiles * 8 ntiles = 64 v_wmma per wave.
// ---------------------------------------------------------------------------
template <bool RELU, bool OUT_HALF>
__global__ __launch_bounds__(256) void sage_gemm_kernel(
    const _Float16* __restrict__ Aself,
    const _Float16* __restrict__ Aneigh,
    const _Float16* __restrict__ wblob,   // [2][32][512] halfs = 64KB
    const float* __restrict__ bias,
    void* __restrict__ outp)
{
    __shared__ _Float16 lds_w[2 * 16384];
    {   // cooperative 64KB load: 256 threads * 16 * 16B
        const v4f* g = (const v4f*)wblob;
        v4f* s = (v4f*)lds_w;
        int t = threadIdx.x;
#pragma unroll
        for (int i = 0; i < 16; ++i) s[t + i * 256] = g[t + i * 256];
    }
    __syncthreads();

    int wave  = threadIdx.x >> 5;
    int lane  = threadIdx.x & 31;
    int mtile = blockIdx.x * 8 + wave;
    if (mtile >= M_TILES) return;          // uniform per wave: EXEC stays all-1s

    int row = mtile * 16 + (lane & 15);
    int hi  = lane >> 4;                   // half-wave selector

    v8f acc[8] = {};
    const _Float16* Aarr[2] = {Aself, Aneigh};

#pragma unroll
    for (int w = 0; w < 2; ++w) {
        const _Float16* A  = Aarr[w];
        const _Float16* wl = lds_w + w * 16384;
#pragma unroll
        for (int kt = 0; kt < 4; ++kt) {
            // A fragment: elems 0-15 = K=kt*32+hi*16+0..15
            int koff = kt * 32 + hi * 16;
            v8h a0 = *(const v8h*)(A + row * 128 + koff);
            v8h a1 = *(const v8h*)(A + row * 128 + koff + 8);
            v16h a;
#pragma unroll
            for (int i = 0; i < 8; ++i) { a[i] = a0[i]; a[8 + i] = a1[i]; }
#pragma unroll
            for (int nt = 0; nt < 8; ++nt) {
                v16h b = *(const v16h*)(wl + (kt * 8 + nt) * 512 + lane * 16);
                acc[nt] = __builtin_amdgcn_wmma_f32_16x16x32_f16(
                    false, a, false, b, (short)0, acc[nt], false, false);
            }
        }
    }

    // Epilogue: D elem i -> row mtile*16 + hi*8 + i, col nt*16 + (lane&15)
#pragma unroll
    for (int nt = 0; nt < 8; ++nt) {
        int col = nt * 16 + (lane & 15);
        float bv = bias[col];
#pragma unroll
        for (int i = 0; i < 8; ++i) {
            float v = acc[nt][i] + bv;
            if (RELU) v = v > 0.0f ? v : 0.0f;
            int r = mtile * 16 + hi * 8 + i;
            if (OUT_HALF) ((_Float16*)outp)[r * 128 + col] = (_Float16)v;
            else          ((float*)outp)[r * 128 + col] = v;
        }
    }
}

// ---------------------------------------------------------------------------
extern "C" void kernel_launch(void* const* d_in, const int* in_sizes, int n_in,
                              void* d_out, int out_size, void* d_ws, size_t ws_size,
                              hipStream_t stream)
{
    const float* client_table = (const float*)d_in[0];
    const float* item_table   = (const float*)d_in[1];
    const float* W_self1      = (const float*)d_in[2];
    const float* W_neigh1     = (const float*)d_in[3];
    const float* b1           = (const float*)d_in[4];
    const float* W_self2      = (const float*)d_in[5];
    const float* W_neigh2     = (const float*)d_in[6];
    const float* b2           = (const float*)d_in[7];
    const int*   node_ids     = (const int*)d_in[8];
    const int*   src          = (const int*)d_in[9];
    const int*   dst          = (const int*)d_in[10];
    float* out = (float*)d_out;

    // Workspace layout (16B-aligned slabs), total ~128.6 MB
    char* ws = (char*)d_ws;
    float*    agg   = (float*)(ws);                        // 51,200,000 B
    _Float16* xh    = (_Float16*)(ws + 51200000);          // 25,600,000 B
    _Float16* hh    = (_Float16*)(ws + 76800000);          // 25,600,000 B
    _Float16* hnh   = (_Float16*)(ws + 102400000);         // 25,600,000 B
    _Float16* wblob = (_Float16*)(ws + 128000000);         //    131,072 B
    float*    deg   = (float*)(ws + 128131072);            //    400,000 B

    const int nodeVec = N_NODES * 32;   // 3.2M threads
    const int edgeVec = N_EDGES * 32;   // 51.2M threads

    // 0) weight repack (all 4 matrices -> fragment-packed f16)
    prep_w_kernel<<<(4 * 16384) / 256, 256, 0, stream>>>(
        W_self1, W_neigh1, W_self2, W_neigh2, wblob);

    // 1) embedding gather -> xh ; zero agg, deg
    embed_kernel<<<nodeVec / 256, 256, 0, stream>>>(
        client_table, item_table, node_ids, xh, agg, deg);

    // 2) degrees
    degree_kernel<<<(N_EDGES + 255) / 256, 256, 0, stream>>>(dst, deg);

    // 3) layer-1 neighbor sum
    scatter_kernel<<<edgeVec / 256, 256, 0, stream>>>(src, dst, xh, agg);

    // 4) mean -> hnh ; re-zero agg
    finalize_kernel<<<nodeVec / 256, 256, 0, stream>>>(agg, deg, hnh);

    // 5) h = relu(x@Ws1 + hn@Wn1 + b1) -> f16
    sage_gemm_kernel<true, true><<<(M_TILES + 7) / 8, 256, 0, stream>>>(
        xh, hnh, wblob, b1, (void*)hh);

    // 6) layer-2 neighbor sum
    scatter_kernel<<<edgeVec / 256, 256, 0, stream>>>(src, dst, hh, agg);

    // 7) mean -> hnh (reuse)
    finalize_kernel<<<nodeVec / 256, 256, 0, stream>>>(agg, deg, hnh);

    // 8) out = h@Ws2 + hn@Wn2 + b2 -> f32
    sage_gemm_kernel<false, false><<<(M_TILES + 7) / 8, 256, 0, stream>>>(
        hh, hnh, wblob + 32768, b2, (void*)out);
}
